// DependencyParser_42992622633716
// MI455X (gfx1250) — matrix operand order repaired and checked by reading, not compiled, not run on hardware
//
#include <hip/hip_runtime.h>
#include <math.h>

// ---------------------------------------------------------------------------
// MI455X (gfx1250) dependency-parser forward pass, round 2.
// - All dense math via v_wmma_f32_16x16x32_f16 (wave32), f32 accumulate.
// - GEMM operands live as f16 in global memory (weights converted once,
//   activations emitted as f16 by their producers), so LDS tiles are f16 and
//   fragment loads are pure contiguous ds_load_b128 with no cvt in the loop.
// - Global->LDS staging via gfx1250 async-to-LDS, double-buffered with a
//   partial s_wait_asynccnt threshold so tile k+1 DMA overlaps tile k WMMAs.
// ---------------------------------------------------------------------------

typedef __attribute__((ext_vector_type(16))) _Float16 v16h;
typedef __attribute__((ext_vector_type(8)))  _Float16 v8h;
typedef __attribute__((ext_vector_type(8)))  float    v8f;

#define DEV __device__ __forceinline__

// Low 32 bits of a flat pointer into the LDS aperture are the LDS byte offset
// (CDNA5 ISA 10.2).
DEV unsigned lds_off_of(const void* p) {
  return (unsigned)(unsigned long long)(size_t)p;
}

// Per-lane 16B async copy global -> LDS (GLOBAL_LOAD_ASYNC_TO_LDS_B128).
DEV void async_ld_b128(unsigned lds_off, const void* gptr) {
  asm volatile("global_load_async_to_lds_b128 %0, %1, off"
               :
               : "v"(lds_off), "v"((unsigned long long)(size_t)gptr)
               : "memory");
}
template <int N>
DEV void wait_async_le() {
  asm volatile("s_wait_asynccnt %0" ::"i"(N) : "memory");
}

// ---------------------------------------------------------------------------
// Batched f16 GEMM:  C[z] = act( A[z] (MxK) * B[z]^T + bias ),  B given (N,K).
// 8 wave32 waves; wave tile = (16*MI) x (16*NJ); block = (32*WR*MI? no:)
//   BM = WR*16*MI rows, BN = WC*16*NJ cols, K stepped by 32.
// z = blockIdx.z decomposed z1 = z/zdiv, z2 = z%zdiv, independent strides.
// Output element (m,n) at C + m*ldcr + n*ldcc (f32 or f16 per OUTH).
// BMODE: 0 none, 1 bias[n] (+bias2[n]), 2 bias[z2].
// Requires M%BM==0, N%BN==0, K%32==0 (true for every shape here), operand
// rows 16B-aligned (all strides are multiples of 8 halves).
// ---------------------------------------------------------------------------
template <int WR, int WC, int MI, int NJ, int BMODE, bool RELU, bool OUTH>
__global__ __launch_bounds__(256) void gemm_f16_wmma(
    const _Float16* __restrict__ A, long lda, long sAz1, long sAz2,
    const _Float16* __restrict__ Bm, long ldb, long sBz1, long sBz2,
    void* __restrict__ Cv, long ldcr, long ldcc, long sCz1, long sCz2,
    const float* __restrict__ bias, const float* __restrict__ bias2,
    int K, int zdiv) {
  static_assert(WR * WC == 8, "8 waves per workgroup");
  constexpr int BM = WR * 16 * MI;
  constexpr int BN = WC * 16 * NJ;
  constexpr int ACH = BM * 4;  // 16B chunks per K-step (32 halves/row)
  constexpr int BCH = BN * 4;
  constexpr bool UNIF = (ACH % 256 == 0) && (BCH % 256 == 0);
  constexpr int WAITN = UNIF ? (ACH + BCH) / 256 : 0;  // per-wave next-batch ops

  __shared__ __align__(16) _Float16 As[2][BM * 32];  // [m][k] f16
  __shared__ __align__(16) _Float16 Bs[2][BN * 32];  // [n][k] f16

  const int tid  = threadIdx.x;
  const int lane = tid & 31;
  const int w    = tid >> 5;
  const int wrow = w % WR;
  const int wcol = w / WR;
  const int half = lane >> 4;

  const int z  = blockIdx.z;
  const int z1 = z / zdiv;
  const int z2 = z % zdiv;

  const long m0 = (long)blockIdx.y * BM;
  const long n0 = (long)blockIdx.x * BN;

  const _Float16* Ab = A + (long)z1 * sAz1 + (long)z2 * sAz2;
  const _Float16* Bb = Bm + (long)z1 * sBz1 + (long)z2 * sBz2;

  v8f acc[MI][NJ];
#pragma unroll
  for (int i = 0; i < MI; ++i)
#pragma unroll
    for (int j = 0; j < NJ; ++j)
#pragma unroll
      for (int v = 0; v < 8; ++v) acc[i][j][v] = 0.0f;

  auto stage = [&](int buf, int k0) {
    for (int c = tid; c < ACH; c += 256) {
      const int row = c >> 2, part = c & 3;
      async_ld_b128(lds_off_of(&As[buf][row * 32 + part * 8]),
                    Ab + (m0 + row) * lda + k0 + part * 8);
    }
    for (int c = tid; c < BCH; c += 256) {
      const int row = c >> 2, part = c & 3;
      async_ld_b128(lds_off_of(&Bs[buf][row * 32 + part * 8]),
                    Bb + (n0 + row) * ldb + k0 + part * 8);
    }
  };

  stage(0, 0);
  int buf = 0;
  for (int k0 = 0; k0 < K; k0 += 32) {
    if (k0 + 32 < K) {
      stage(buf ^ 1, k0 + 32);       // overlap next tile's DMA with this WMMA
      wait_async_le<WAITN>();        // drain only the current tile's copies
    } else {
      wait_async_le<0>();
    }
    __syncthreads();

    // Fragment loads: pure contiguous 16B ds reads, no conversion.
    // A (ISA 7.12.2): lane m = lane&15; e0..7 <- k=8h.., e8..15 <- k=16+8h..
    v16h afr[MI], bfr[NJ];
#pragma unroll
    for (int i = 0; i < MI; ++i) {
      const int m = wrow * (16 * MI) + i * 16 + (lane & 15);
      const _Float16* r = &As[buf][m * 32];
      const v8h lo = *(const v8h*)(r + 8 * half);
      const v8h hi = *(const v8h*)(r + 16 + 8 * half);
      v16h a;
#pragma unroll
      for (int e = 0; e < 8; ++e) { a[e] = lo[e]; a[8 + e] = hi[e]; }
      afr[i] = a;
    }
    // B: lane n = lane&15; e0..7 <- k=8h.., e8..15 <- k=16+8h.. (same as A)
#pragma unroll
    for (int j = 0; j < NJ; ++j) {
      const int n = wcol * (16 * NJ) + j * 16 + (lane & 15);
      const _Float16* r = &Bs[buf][n * 32];
      const v8h lo = *(const v8h*)(r + 8 * half);
      const v8h hi = *(const v8h*)(r + 16 + 8 * half);
      v16h b;
#pragma unroll
      for (int e = 0; e < 8; ++e) { b[e] = lo[e]; b[8 + e] = hi[e]; }
      bfr[j] = b;
    }
#pragma unroll
    for (int i = 0; i < MI; ++i)
#pragma unroll
      for (int j = 0; j < NJ; ++j)
        acc[i][j] = __builtin_amdgcn_wmma_f32_16x16x32_f16(
            false, afr[i], false, bfr[j], (short)0, acc[i][j], false, false);
    __syncthreads();  // protect buf^1 from being restaged while others read
    buf ^= 1;
  }

  float*    Cf = (float*)Cv;
  _Float16* Ch = (_Float16*)Cv;
  const long cz = (long)z1 * sCz1 + (long)z2 * sCz2;
#pragma unroll
  for (int i = 0; i < MI; ++i) {
#pragma unroll
    for (int j = 0; j < NJ; ++j) {
      const long nf    = n0 + wcol * (16 * NJ) + j * 16 + (lane & 15);
      const long mbase = m0 + wrow * (16 * MI) + i * 16 + 8 * half;
      float bv = 0.0f;
      if (BMODE == 1) { bv = bias[nf]; if (bias2) bv += bias2[nf]; }
      if (BMODE == 2) { bv = bias[z2]; }
#pragma unroll
      for (int v = 0; v < 8; ++v) {
        float val = acc[i][j][v] + bv;
        if (RELU) val = fmaxf(val, 0.0f);
        const long idx = cz + (mbase + v) * ldcr + nf * ldcc;
        if (OUTH) Ch[idx] = (_Float16)val; else Cf[idx] = val;
      }
    }
  }
}

template <int WR, int WC, int MI, int NJ, int BMODE, bool RELU, bool OUTH>
static inline void launch_gemm(hipStream_t st, int M, int N, int Z,
                               const _Float16* A, long lda, long sA1, long sA2,
                               const _Float16* B, long ldb, long sB1, long sB2,
                               void* C, long ldcr, long ldcc, long sC1, long sC2,
                               const float* bias, const float* bias2, int K,
                               int zdiv) {
  dim3 g(N / (WC * 16 * NJ), M / (WR * 16 * MI), Z);
  gemm_f16_wmma<WR, WC, MI, NJ, BMODE, RELU, OUTH><<<g, 256, 0, st>>>(
      A, lda, sA1, sA2, B, ldb, sB1, sB2, C, ldcr, ldcc, sC1, sC2, bias, bias2,
      K, zdiv);
}

// ---------------------------------------------------------------------------
// Conversion / elementwise kernels
// ---------------------------------------------------------------------------
__global__ void cvt_f32_f16(const float* __restrict__ in,
                            _Float16* __restrict__ out) {
  const long i = (long)blockIdx.x * 256 + threadIdx.x;
  out[i] = (_Float16)in[i];
}

// VT[bh][d][tok] = V part of QKVh  (B operand for P@V, (N=d, K=tok))
__global__ void transpose_v(const _Float16* __restrict__ qkvh,
                            _Float16* __restrict__ vt) {
  const long idx = (long)blockIdx.x * 256 + threadIdx.x;  // 32*96*512
  const int tok = (int)(idx & 511);
  const int d   = (int)((idx >> 9) % 96);
  const int bh  = (int)(idx / (96 * 512));
  const int b = bh >> 3, h = bh & 7;
  vt[idx] = qkvh[((long)b * 512 + tok) * 2304 + 1536 + h * 96 + d];
}

// BWT[r][e][d] = biaf_w[r][d][e]  (B operand for head@biaf_w, (N=e, K=d))
__global__ void transpose_bw(const float* __restrict__ bw,
                             _Float16* __restrict__ bwt) {
  const long idx = (long)blockIdx.x * 256 + threadIdx.x;  // 50*384*384
  const int d = (int)(idx % 384);
  const int e = (int)((idx / 384) % 384);
  const int r = (int)(idx / (384 * 384));
  bwt[idx] = (_Float16)bw[(long)r * 384 * 384 + (long)d * 384 + e];
}

__global__ void embed_kernel(const int* __restrict__ ids,
                             const float* __restrict__ ew,
                             const float* __restrict__ ep,
                             float* __restrict__ out,
                             _Float16* __restrict__ outh) {
  const long idx = (long)blockIdx.x * 256 + threadIdx.x;  // 2048*768
  const int  d   = (int)(idx % 768);
  const long bl  = idx / 768;
  const int  l   = (int)(bl % 512);
  const float v = ew[(long)ids[bl] * 768 + d] + ep[(long)l * 768 + d];
  out[idx]  = v;
  outh[idx] = (_Float16)v;
}

__global__ __launch_bounds__(256) void ln_residual(
    const float* __restrict__ x, const float* __restrict__ y,
    const float* __restrict__ g, const float* __restrict__ b,
    float* __restrict__ out, _Float16* __restrict__ outh) {
  const long row = blockIdx.x;  // 2048 rows of 768
  const float* xr = x + row * 768;
  const float* yr = y + row * 768;
  float v[3], s = 0.f, s2 = 0.f;
#pragma unroll
  for (int i = 0; i < 3; ++i) {
    const int c = threadIdx.x + i * 256;
    const float t = xr[c] + yr[c];
    v[i] = t; s += t; s2 += t * t;
  }
  for (int o = 16; o; o >>= 1) { s += __shfl_xor(s, o, 32); s2 += __shfl_xor(s2, o, 32); }
  __shared__ float sh0[8], sh1[8];
  const int wv = threadIdx.x >> 5, lane = threadIdx.x & 31;
  if (lane == 0) { sh0[wv] = s; sh1[wv] = s2; }
  __syncthreads();
  if (threadIdx.x == 0) {
    float a = 0.f, a2 = 0.f;
    for (int i = 0; i < 8; ++i) { a += sh0[i]; a2 += sh1[i]; }
    sh0[0] = a; sh1[0] = a2;
  }
  __syncthreads();
  const float mean = sh0[0] * (1.0f / 768.0f);
  const float var  = sh1[0] * (1.0f / 768.0f) - mean * mean;
  const float inv  = rsqrtf(var + 1e-5f);
#pragma unroll
  for (int i = 0; i < 3; ++i) {
    const int c = threadIdx.x + i * 256;
    const float r = (v[i] - mean) * inv * g[c] + b[c];
    out[row * 768 + c]  = r;
    outh[row * 768 + c] = (_Float16)r;
  }
}

// softmax over rows of 512; reads f32 scores, writes f16 probabilities.
__global__ __launch_bounds__(256) void softmax512(const float* __restrict__ S,
                                                  _Float16* __restrict__ P,
                                                  float scale) {
  const float* row = S + (size_t)blockIdx.x * 512;
  _Float16*    po  = P + (size_t)blockIdx.x * 512;
  const int tid = threadIdx.x;
  const float a = row[tid] * scale;
  const float b = row[tid + 256] * scale;
  __shared__ float sh[8];
  float m = fmaxf(a, b);
  for (int o = 16; o; o >>= 1) m = fmaxf(m, __shfl_xor(m, o, 32));
  if ((tid & 31) == 0) sh[tid >> 5] = m;
  __syncthreads();
  if (tid == 0) { float t = sh[0]; for (int i = 1; i < 8; ++i) t = fmaxf(t, sh[i]); sh[0] = t; }
  __syncthreads();
  m = sh[0];
  const float ea = __expf(a - m), eb = __expf(b - m);
  float s = ea + eb;
  __syncthreads();
  for (int o = 16; o; o >>= 1) s += __shfl_xor(s, o, 32);
  if ((tid & 31) == 0) sh[tid >> 5] = s;
  __syncthreads();
  if (tid == 0) { float t = 0.f; for (int i = 0; i < 8; ++i) t += sh[i]; sh[0] = t; }
  __syncthreads();
  const float inv = 1.0f / sh[0];
  po[tid]       = (_Float16)(ea * inv);
  po[tid + 256] = (_Float16)(eb * inv);
}

// Persistent bidirectional LSTM scan: grid.x = 2 (0 fwd, 1 bwd), 512 threads.
// Recurrence is latency-bound (M=4), so state stays in LDS and the gate
// matvec is VALU; output h is written as f16 (only GEMMs consume it).
__global__ __launch_bounds__(512) void lstm_scan(
    const float* __restrict__ gx_f, const float* __restrict__ gx_b,
    const float* __restrict__ whh_f, const float* __restrict__ whh_b,
    _Float16* __restrict__ hout) {
  const int dir = blockIdx.x;
  const float* gx  = dir ? gx_b : gx_f;
  const float* whh = dir ? whh_b : whh_f;
  __shared__ float hs[4][384];
  __shared__ float cs[4][384];
  __shared__ float gs[4][1536];
  const int tid = threadIdx.x;
  for (int i = tid; i < 4 * 384; i += 512) { hs[i / 384][i % 384] = 0.f; cs[i / 384][i % 384] = 0.f; }
  __syncthreads();
  for (int step = 0; step < 512; ++step) {
    const int t = dir ? (511 - step) : step;
    for (int n = tid; n < 1536; n += 512) {
      float a0 = 0.f, a1 = 0.f, a2 = 0.f, a3 = 0.f;
      const float* wr = whh + (long)n * 384;
      for (int k = 0; k < 384; ++k) {
        const float wv = wr[k];
        a0 += wv * hs[0][k]; a1 += wv * hs[1][k];
        a2 += wv * hs[2][k]; a3 += wv * hs[3][k];
      }
      gs[0][n] = a0 + gx[((long)0 * 512 + t) * 1536 + n];
      gs[1][n] = a1 + gx[((long)1 * 512 + t) * 1536 + n];
      gs[2][n] = a2 + gx[((long)2 * 512 + t) * 1536 + n];
      gs[3][n] = a3 + gx[((long)3 * 512 + t) * 1536 + n];
    }
    __syncthreads();
    for (int idx = tid; idx < 4 * 384; idx += 512) {
      const int b = idx / 384, j = idx % 384;
      const float ig = 1.0f / (1.0f + __expf(-gs[b][j]));
      const float fg = 1.0f / (1.0f + __expf(-gs[b][384 + j]));
      const float gg = tanhf(gs[b][768 + j]);
      const float og = 1.0f / (1.0f + __expf(-gs[b][1152 + j]));
      const float c  = fg * cs[b][j] + ig * gg;
      const float h  = og * tanhf(c);
      cs[b][j] = c; hs[b][j] = h;
      hout[((long)b * 512 + t) * 768 + dir * 384 + j] = (_Float16)h;
    }
    __syncthreads();
  }
}

// out[row] = dot(feat[row,:384], w) + bias[0]; one wave per row.
__global__ __launch_bounds__(256) void arc_proj(
    const _Float16* __restrict__ feat, const float* __restrict__ w,
    const float* __restrict__ bias, float* __restrict__ out) {
  const int row  = blockIdx.x * 8 + (threadIdx.x >> 5);
  const int lane = threadIdx.x & 31;
  const _Float16* fr = feat + (long)row * 384;
  float s = 0.f;
  for (int k = lane; k < 384; k += 32) s += (float)fr[k] * w[k];
  for (int o = 16; o; o >>= 1) s += __shfl_xor(s, o, 32);
  if (lane == 0) out[row] = s + bias[0];
}

__global__ void arc_combine(const float* __restrict__ ah,
                            const float* __restrict__ ad,
                            float* __restrict__ out) {
  const long idx = (long)blockIdx.x * 256 + threadIdx.x;  // 4*512*512
  const int  j   = (int)(idx & 511);
  const long bi  = idx >> 9;
  const int  b   = (int)(bi >> 9);
  out[idx] = ah[bi] + ad[(long)b * 512 + j];
}

__global__ __launch_bounds__(256) void argmax512(const float* __restrict__ S,
                                                 float* __restrict__ out) {
  const float* row = S + (size_t)blockIdx.x * 512;
  const int tid = threadIdx.x;
  float v1 = row[tid], v2 = row[tid + 256];
  float bv; int bi;
  if (v2 > v1) { bv = v2; bi = tid + 256; } else { bv = v1; bi = tid; }
  __shared__ float sv[256];
  __shared__ int   si[256];
  sv[tid] = bv; si[tid] = bi;
  __syncthreads();
  for (int s = 128; s; s >>= 1) {
    if (tid < s) {
      const float ov = sv[tid + s]; const int oi = si[tid + s];
      if (ov > sv[tid] || (ov == sv[tid] && oi < si[tid])) { sv[tid] = ov; si[tid] = oi; }
    }
    __syncthreads();
  }
  if (tid == 0) out[blockIdx.x] = (float)si[0];
}

__global__ void argmax50(const float* __restrict__ rel,
                         float* __restrict__ out, long nrows) {
  const long r = (long)blockIdx.x * 256 + threadIdx.x;
  if (r >= nrows) return;
  const float* p = rel + r * 50;
  float bv = p[0]; int bi = 0;
  for (int i = 1; i < 50; ++i) if (p[i] > bv) { bv = p[i]; bi = i; }
  out[r] = (float)bi;
}

// ---------------------------------------------------------------------------
// Host orchestration
// ---------------------------------------------------------------------------
extern "C" void kernel_launch(void* const* d_in, const int* in_sizes, int n_in,
                              void* d_out, int out_size, void* d_ws,
                              size_t ws_size, hipStream_t stream) {
  (void)in_sizes; (void)n_in; (void)out_size; (void)ws_size;
  const int*   ids       = (const int*)d_in[0];
  const float* emb_word  = (const float*)d_in[1];
  const float* emb_pos   = (const float*)d_in[2];
  const float* t_in_w    = (const float*)d_in[3];
  const float* t_in_b    = (const float*)d_in[4];
  const float* t_out_w   = (const float*)d_in[5];
  const float* t_out_b   = (const float*)d_in[6];
  const float* t_ff1_w   = (const float*)d_in[7];
  const float* t_ff1_b   = (const float*)d_in[8];
  const float* t_ff2_w   = (const float*)d_in[9];
  const float* t_ff2_b   = (const float*)d_in[10];
  const float* t_ln1_g   = (const float*)d_in[11];
  const float* t_ln1_b   = (const float*)d_in[12];
  const float* t_ln2_g   = (const float*)d_in[13];
  const float* t_ln2_b   = (const float*)d_in[14];
  const float* lstm_wih  = (const float*)d_in[15];
  const float* lstm_whh  = (const float*)d_in[16];
  const float* lstm_bih  = (const float*)d_in[17];
  const float* lstm_bhh  = (const float*)d_in[18];
  const float* head_w    = (const float*)d_in[19];
  const float* head_b    = (const float*)d_in[20];
  const float* dep_w     = (const float*)d_in[21];
  const float* dep_b     = (const float*)d_in[22];
  const float* arc_head_w = (const float*)d_in[23];
  const float* arc_head_b = (const float*)d_in[24];
  const float* arc_dep_w  = (const float*)d_in[25];
  const float* arc_dep_b  = (const float*)d_in[26];
  const float* biaf_w    = (const float*)d_in[27];
  const float* biaf_b    = (const float*)d_in[28];

  // ---- workspace carve-out (f32 region then f16 region; ~200 MB total) ----
  float* ws = (float*)d_ws;
  size_t off = 0;
  auto carveF = [&](size_t n) { float* p = ws + off; off += n; return p; };
  float* X    = carveF((size_t)2048 * 768);
  float* X2   = carveF((size_t)2048 * 768);
  float* S    = carveF((size_t)32 * 512 * 512);
  float* Y    = carveF((size_t)2048 * 768);
  float* GXF  = carveF((size_t)2048 * 1536);
  float* GXB  = carveF((size_t)2048 * 1536);
  float* AH   = carveF((size_t)2048);
  float* AD   = carveF((size_t)2048);

  _Float16* hbase = (_Float16*)(ws + off);
  size_t hoff = 0;
  auto carveH = [&](size_t n) { _Float16* p = hbase + hoff; hoff += n; return p; };
  _Float16* Xh    = carveH((size_t)2048 * 768);
  _Float16* X2h   = carveH((size_t)2048 * 768);
  _Float16* Hh0   = carveH((size_t)2048 * 768);
  _Float16* Hh1   = carveH((size_t)2048 * 768);
  _Float16* QKVh  = carveH((size_t)2048 * 2304);
  _Float16* Ph    = carveH((size_t)32 * 512 * 512);
  _Float16* Oh    = carveH((size_t)2048 * 768);
  _Float16* FFh   = carveH((size_t)2048 * 3072);
  _Float16* VTh   = carveH((size_t)32 * 96 * 512);
  _Float16* HEADh = carveH((size_t)2048 * 384);
  _Float16* DEPh  = carveH((size_t)2048 * 384);
  _Float16* TMPh  = carveH((size_t)512 * 50 * 384);
  // reusable per-layer f16 weight buffers
  _Float16* WinH  = carveH((size_t)2304 * 768);
  _Float16* WoutH = carveH((size_t)768 * 768);
  _Float16* Wf1H  = carveH((size_t)3072 * 768);
  _Float16* Wf2H  = carveH((size_t)768 * 3072);
  _Float16* WihH  = carveH((size_t)2 * 1536 * 768);
  _Float16* WhdH  = carveH((size_t)2 * 384 * 768);  // head_w then dep_w
  _Float16* BWTh  = carveH((size_t)50 * 384 * 384);

  float* out_arc     = (float*)d_out;              // 4*512*512
  float* out_rel     = out_arc + (size_t)1048576;  // 4*512*512*50
  float* out_arc_idx = out_rel + (size_t)52428800; // 4*512
  float* out_rel_idx = out_arc_idx + 2048;         // 4*512*512

  // 1) embeddings (f32 + f16)
  embed_kernel<<<6144, 256, 0, stream>>>(ids, emb_word, emb_pos, X, Xh);

  // 2) transformer layers
  for (int l = 0; l < 3; ++l) {
    cvt_f32_f16<<<(2304 * 768) / 256, 256, 0, stream>>>(t_in_w + (size_t)l * 2304 * 768, WinH);
    cvt_f32_f16<<<(768 * 768) / 256, 256, 0, stream>>>(t_out_w + (size_t)l * 768 * 768, WoutH);
    cvt_f32_f16<<<(3072 * 768) / 256, 256, 0, stream>>>(t_ff1_w + (size_t)l * 3072 * 768, Wf1H);
    cvt_f32_f16<<<(768 * 3072) / 256, 256, 0, stream>>>(t_ff2_w + (size_t)l * 768 * 3072, Wf2H);
    const float* inB = t_in_b + (size_t)l * 2304;
    const float* oB  = t_out_b + (size_t)l * 768;
    const float* f1B = t_ff1_b + (size_t)l * 3072;
    const float* f2B = t_ff2_b + (size_t)l * 768;

    // QKVh = f16( Xh @ WinH^T + inB )
    launch_gemm<4, 2, 2, 4, 1, false, true>(stream, 2048, 2304, 1,
        Xh, 768, 0, 0, WinH, 768, 0, 0,
        QKVh, 2304, 1, 0, 0, inB, nullptr, 768, 1);

    // S[b,h] = Q @ K^T  (z = b*8+h)
    launch_gemm<4, 2, 2, 4, 0, false, false>(stream, 512, 512, 32,
        QKVh, 2304, (long)512 * 2304, 96,
        QKVh + 768, 2304, (long)512 * 2304, 96,
        S, 512, 1, (long)8 * 512 * 512, (long)512 * 512,
        nullptr, nullptr, 96, 8);

    softmax512<<<16384, 256, 0, stream>>>(S, Ph, 0.1020620726159658f);
    transpose_v<<<6144, 256, 0, stream>>>(QKVh, VTh);

    // Oh[b,:,h*96:] = P @ V   (N=96 -> 256x32 tile config)
    launch_gemm<8, 1, 2, 2, 0, false, true>(stream, 512, 96, 32,
        Ph, 512, (long)8 * 512 * 512, (long)512 * 512,
        VTh, 512, (long)8 * 96 * 512, (long)96 * 512,
        Oh, 768, 1, (long)512 * 768, 96,
        nullptr, nullptr, 512, 8);

    // Y = Oh @ WoutH^T + oB   (f32 for LN)
    launch_gemm<4, 2, 2, 4, 1, false, false>(stream, 2048, 768, 1,
        Oh, 768, 0, 0, WoutH, 768, 0, 0,
        Y, 768, 1, 0, 0, oB, nullptr, 768, 1);

    ln_residual<<<2048, 256, 0, stream>>>(X, Y, t_ln1_g + l * 768, t_ln1_b + l * 768, X2, X2h);

    // FFh = relu(X2h @ Wf1H^T + f1B)
    launch_gemm<4, 2, 2, 4, 1, true, true>(stream, 2048, 3072, 1,
        X2h, 768, 0, 0, Wf1H, 768, 0, 0,
        FFh, 3072, 1, 0, 0, f1B, nullptr, 768, 1);

    // Y = FFh @ Wf2H^T + f2B
    launch_gemm<4, 2, 2, 4, 1, false, false>(stream, 2048, 768, 1,
        FFh, 3072, 0, 0, Wf2H, 3072, 0, 0,
        Y, 768, 1, 0, 0, f2B, nullptr, 3072, 1);

    ln_residual<<<2048, 256, 0, stream>>>(X2, Y, t_ln2_g + l * 768, t_ln2_b + l * 768, X, Xh);
  }

  // 3) BiLSTM stack (gx GEMMs in f16, recurrence in f32 LDS)
  const _Float16* HcurH = Xh;
  _Float16* Hbufs[2] = {Hh0, Hh1};
  for (int l = 0; l < 3; ++l) {
    cvt_f32_f16<<<(2 * 1536 * 768) / 256, 256, 0, stream>>>(
        lstm_wih + (size_t)l * 2 * 1536 * 768, WihH);
    launch_gemm<4, 2, 2, 4, 1, false, false>(stream, 2048, 1536, 1,
        HcurH, 768, 0, 0, WihH, 768, 0, 0,
        GXF, 1536, 1, 0, 0,
        lstm_bih + (size_t)(l * 2 + 0) * 1536,
        lstm_bhh + (size_t)(l * 2 + 0) * 1536, 768, 1);
    launch_gemm<4, 2, 2, 4, 1, false, false>(stream, 2048, 1536, 1,
        HcurH, 768, 0, 0, WihH + (size_t)1536 * 768, 768, 0, 0,
        GXB, 1536, 1, 0, 0,
        lstm_bih + (size_t)(l * 2 + 1) * 1536,
        lstm_bhh + (size_t)(l * 2 + 1) * 1536, 768, 1);
    lstm_scan<<<2, 512, 0, stream>>>(
        GXF, GXB,
        lstm_whh + (size_t)(l * 2 + 0) * 1536 * 384,
        lstm_whh + (size_t)(l * 2 + 1) * 1536 * 384,
        Hbufs[l & 1]);
    HcurH = Hbufs[l & 1];
  }

  // 4) head / dep MLPs (f16 out; consumed only by GEMMs and arc_proj)
  cvt_f32_f16<<<(384 * 768) / 256, 256, 0, stream>>>(head_w, WhdH);
  cvt_f32_f16<<<(384 * 768) / 256, 256, 0, stream>>>(dep_w, WhdH + (size_t)384 * 768);
  launch_gemm<4, 2, 2, 4, 1, true, true>(stream, 2048, 384, 1,
      HcurH, 768, 0, 0, WhdH, 768, 0, 0,
      HEADh, 384, 1, 0, 0, head_b, nullptr, 768, 1);
  launch_gemm<4, 2, 2, 4, 1, true, true>(stream, 2048, 384, 1,
      HcurH, 768, 0, 0, WhdH + (size_t)384 * 768, 768, 0, 0,
      DEPh, 384, 1, 0, 0, dep_b, nullptr, 768, 1);

  // 5) arc scores
  arc_proj<<<256, 256, 0, stream>>>(HEADh, arc_head_w, arc_head_b, AH);
  arc_proj<<<256, 256, 0, stream>>>(DEPh, arc_dep_w, arc_dep_b, AD);
  arc_combine<<<4096, 256, 0, stream>>>(AH, AD, out_arc);

  // 6) biaffine relation scores (per batch to bound workspace)
  transpose_bw<<<(50 * 384 * 384) / 256, 256, 0, stream>>>(biaf_w, BWTh);
  for (int b = 0; b < 4; ++b) {
    // TMPh[i,r,e] = HEAD_b[i,:] @ biaf_w[r]   (z = r)
    launch_gemm<4, 2, 2, 4, 0, false, true>(stream, 512, 384, 50,
        HEADh + (size_t)b * 512 * 384, 384, 0, 0,
        BWTh, 384, 0, (long)384 * 384,
        TMPh, (long)50 * 384, 1, 0, 384,
        nullptr, nullptr, 384, 50);
    // rel[b,i,j,r] = TMPh[i,r,:] @ DEP_b[j,:]^T + biaf_b[r]
    launch_gemm<4, 2, 2, 4, 2, false, false>(stream, 512, 512, 50,
        TMPh, (long)50 * 384, 0, 384,
        DEPh + (size_t)b * 512 * 384, 384, 0, 0,
        out_rel + (size_t)b * 512 * 512 * 50, (long)512 * 50, 50, 0, 1,
        biaf_b, nullptr, 384, 50);
  }

  // 7) argmax outputs (d_out is f32; indices stored as floats)
  argmax512<<<2048, 256, 0, stream>>>(out_arc, out_arc_idx);
  argmax50<<<4096, 256, 0, stream>>>(out_rel, out_rel_idx, (long)1048576);
}
